// Model_15951508538244
// MI455X (gfx1250) — matrix operand order repaired and checked by reading, not compiled, not run on hardware
//
#include <hip/hip_runtime.h>

// ---------------------------------------------------------------------------
// MI455X (gfx1250) fused implementation.
//
// 4 batched 2048x2048x128 similarity GEMMs (~17.2 GFLOP) in f16 WMMA with f32
// accum; score matrices never hit HBM (top-k/softmax/gather fused).  The WMMA
// kernel is software-pipelined: B fragments are register-double-buffered so
// global loads for tile t+1 issue before tile t's v_wmma chain, and the
// VALU-heavy top-k insertion for tile t-1 runs while tile t's 8 WMMAs drain
// in the matrix pipe (XDL WMMA co-executes with VALU).  32 rows per wave ->
// every lane owns a top-k row and B traffic is halved; per-batch f16 operand
// (512 KB) lives in the 192 MB L2.  All hot code is macro-inlined (round-2
// showed clang outlining the top-k lambda into a call with scratch spills).
// ---------------------------------------------------------------------------

typedef __attribute__((ext_vector_type(16))) _Float16 v16h;
typedef __attribute__((ext_vector_type(8)))  _Float16 v8h;
typedef __attribute__((ext_vector_type(8)))  float    v8f;
typedef __attribute__((ext_vector_type(4)))  float    v4f;

namespace {

constexpr int B_ = 4;
constexpr int N_ = 2048;
constexpr int F_ = 128;
constexpr int K_CROSS = 10;          // == K_SELF
constexpr float INV_HEAT = 1.0f / 8.0f;

// ---------- unit-normalize feature rows, emit f16 (WMMA operand dtype) ------
__global__ __launch_bounds__(256)
void normalize_f16_kernel(const float* __restrict__ feat, _Float16* __restrict__ out)
{
    const int row  = blockIdx.x * 8 + (threadIdx.x >> 5);   // 8 rows / block (wave32)
    const int lane = threadIdx.x & 31;
    const float* fr = feat + (size_t)row * F_;
    const float v0 = fr[lane], v1 = fr[lane + 32], v2 = fr[lane + 64], v3 = fr[lane + 96];
    float s = v0 * v0 + v1 * v1 + v2 * v2 + v3 * v3;
#pragma unroll
    for (int off = 16; off >= 1; off >>= 1) s += __shfl_xor(s, off, 32);
    const float r = rsqrtf(s);
    _Float16* orow = out + (size_t)row * F_;
    orow[lane]      = (_Float16)(v0 * r);
    orow[lane + 32] = (_Float16)(v1 * r);
    orow[lane + 64] = (_Float16)(v2 * r);
    orow[lane + 96] = (_Float16)(v3 * r);
}

// ---------- fused cos-sim GEMM + row top-K + softmax + 3D gather ------------
// One wave handles 32 query rows (two 16x128 A fragment sets).  Loop over 128
// key tiles: register-double-buffered B fragments, two independent WMMA accum
// chains, top-k scan of the previous tile overlapped with current WMMAs via a
// double-buffered LDS score tile.  Same-wave LDS ops are in-order (ISA 7.3),
// so no barriers are required in this single-wave workgroup.

#define LOAD_B(bf, t_) do {                                                    \
    const _Float16* _p = Bbase + (size_t)(t_) * (16 * F_);                     \
    __builtin_prefetch(_p + 16 * F_, 0, 1);  /* global_prefetch_b8, next tile */\
    _Pragma("unroll")                                                          \
    for (int _c = 0; _c < 4; ++_c) (bf)[_c] = *(const v16h*)(_p + 32 * _c);    \
} while (0)

#define WMMA_TILE(bf) do {                                                     \
    a0 = vzero; a1 = vzero;                                                    \
    _Pragma("unroll")                                                          \
    for (int _c = 0; _c < 4; ++_c) {                                           \
        a0 = __builtin_amdgcn_wmma_f32_16x16x32_f16(                           \
                 false, afrag[_c],     false, (bf)[_c], (short)0, a0, false, false); \
        a1 = __builtin_amdgcn_wmma_f32_16x16x32_f16(                           \
                 false, afrag[4 + _c], false, (bf)[_c], (short)0, a1, false, false); \
    }                                                                          \
} while (0)

// C/D layout: element e of lane -> row e + 8*hi (chain 0) / 16+e+8*hi (chain 1),
// column lane&15.
#define STORE_SC(t_) do {                                                      \
    float* _dst = &s_sc[(t_) & 1][0][0];                                       \
    _Pragma("unroll")                                                          \
    for (int _e = 0; _e < 8; ++_e) {                                           \
        _dst[(_e + 8 * hi) * 20 + r]      = a0[_e];                            \
        _dst[(16 + _e + 8 * hi) * 20 + r] = a1[_e];                            \
    }                                                                          \
} while (0)

#define TOPK_SCAN(t_) do {                                                     \
    const v4f* _rp = (const v4f*)&s_sc[(t_) & 1][lane][0];                     \
    const int _cb = (t_) * 16;                                                 \
    _Pragma("unroll")                                                          \
    for (int _q = 0; _q < 4; ++_q) {                                           \
        const v4f _quad = _rp[_q];                                             \
        _Pragma("unroll")                                                      \
        for (int _e = 0; _e < 4; ++_e) {                                       \
            const float _v = _quad[_e];                                        \
            if (_v > kv[KT - 1]) {           /* sorted-descending insertion */ \
                kv[KT - 1] = _v; ki[KT - 1] = _cb + 4 * _q + _e;               \
                _Pragma("unroll")                                              \
                for (int _p = KT - 1; _p > 0; --_p) {                          \
                    const bool _sw = kv[_p] > kv[_p - 1];                      \
                    const float _tv = kv[_p]; const int _ti = ki[_p];          \
                    kv[_p]     = _sw ? kv[_p - 1] : _tv;                       \
                    ki[_p]     = _sw ? ki[_p - 1] : _ti;                       \
                    kv[_p - 1] = _sw ? _tv : kv[_p - 1];                       \
                    ki[_p - 1] = _sw ? _ti : ki[_p - 1];                       \
                }                                                              \
            }                                                                  \
        }                                                                      \
    }                                                                          \
} while (0)

template <int KT, int SKIP>
__global__ __launch_bounds__(32, 1)   // 1 wave/WG, allow full VGPR budget
void wmma_topk_recon(const _Float16* __restrict__ Af,
                     const _Float16* __restrict__ Bf,
                     const float*    __restrict__ pts,
                     float*          __restrict__ out)
{
    constexpr int TILES = N_ / 16;
    const int lane = threadIdx.x;                 // 0..31 (wave32)
    const int tilesPerBatch = N_ / 32;            // 64 row-tiles per batch
    const int b  = blockIdx.x / tilesPerBatch;
    const int m0 = (blockIdx.x % tilesPerBatch) * 32;
    const size_t base = (size_t)b * N_;

    const int r  = lane & 15;                     // row (A) / column (B) in 16-tile
    const int hi = lane >> 4;                     // half-wave select
    const int khalf = hi * 16;                    // B layout: halves split K 0..15/16..31

    // A fragments: contiguous K order used for A and B -> lane<16 holds K {0..15} of chunk,
    // lane>=16 holds K {16..31}.  afrag[0..3] rows m0..m0+15,
    // afrag[4..7] rows m0+16..m0+31.
    v16h afrag[8];
#pragma unroll
    for (int h = 0; h < 2; ++h) {
        const _Float16* Arow = Af + (base + m0 + 16 * h + r) * F_;
#pragma unroll
        for (int c = 0; c < 4; ++c) {
            const int kb = 32 * c;
            v8h lo = *(const v8h*)(Arow + kb + hi * 16);
            v8h hh = *(const v8h*)(Arow + kb + 8 + hi * 16);
            afrag[4 * h + c] = __builtin_shufflevector(lo, hh, 0, 1, 2, 3, 4, 5, 6, 7,
                                                               8, 9, 10, 11, 12, 13, 14, 15);
        }
    }

    // per-row top-K registers: lane l owns query row m0+l
    float kv[KT]; int ki[KT];
#pragma unroll
    for (int k = 0; k < KT; ++k) { kv[k] = -3.0e38f; ki[k] = 0; }

    // double-buffered 32x16 score tile; rows padded to 20 floats (16B aligned)
    __shared__ alignas(16) float s_sc[2][32][20];

    const _Float16* Bbase = Bf + (base + r) * F_ + khalf;
    const v8f vzero = {};
    v16h b0f[4], b1f[4];
    v8f a0, a1;

    LOAD_B(b0f, 0);
    for (int t = 0; t < TILES; t += 2) {          // 2x unrolled register double-buffer
        LOAD_B(b1f, t + 1);                       // t+1 < TILES always (TILES even)
        WMMA_TILE(b0f);
        if (t > 0) TOPK_SCAN(t - 1);              // overlaps with WMMA drain
        STORE_SC(t);
        if (t + 2 < TILES) LOAD_B(b0f, t + 2);
        WMMA_TILE(b1f);
        TOPK_SCAN(t);
        STORE_SC(t + 1);
    }
    TOPK_SCAN(TILES - 1);

    // softmax over kept K values + weighted 3D gather (all 32 lanes active)
    const float mx = kv[SKIP];                    // list is sorted descending
    float wsum = 0.f, ax = 0.f, ay = 0.f, az = 0.f;
#pragma unroll
    for (int k = SKIP; k < KT; ++k) {
        const float w = __expf(kv[k] - mx);
        const float* p = pts + (base + ki[k]) * 3;
        wsum += w; ax += w * p[0]; ay += w * p[1]; az += w * p[2];
    }
    const float inv = 1.0f / wsum;
    float* o = out + (base + m0 + lane) * 3;
    o[0] = ax * inv; o[1] = ay * inv; o[2] = az * inv;
}

#undef LOAD_B
#undef WMMA_TILE
#undef STORE_SC
#undef TOPK_SCAN

// ---------- kNN: top-10 nearest (only first 10 of the 27 are ever used) -----
__global__ __launch_bounds__(256)
void knn10_kernel(const float* __restrict__ pos, int* __restrict__ knn)
{
    __shared__ float sx[256], sy[256], sz[256];
    const int tid = threadIdx.x;
    const int gid = blockIdx.x * 256 + tid;
    const int b = gid / N_, n = gid % N_;
    const size_t base = (size_t)b * N_;
    const float* prow = pos + (base + n) * 3;
    const float px = prow[0], py = prow[1], pz = prow[2];

    float kd[K_CROSS]; int kidx[K_CROSS];
#pragma unroll
    for (int k = 0; k < K_CROSS; ++k) { kd[k] = 3.0e38f; kidx[k] = 0; }

    for (int t0 = 0; t0 < N_; t0 += 256) {
        __syncthreads();
        const float* bp = pos + (base + t0 + tid) * 3;
        sx[tid] = bp[0]; sy[tid] = bp[1]; sz[tid] = bp[2];
        __syncthreads();
        for (int j = 0; j < 256; ++j) {
            const float dx = px - sx[j], dy = py - sy[j], dz = pz - sz[j];
            const float d = dx * dx + dy * dy + dz * dz;
            if (d < kd[K_CROSS - 1]) {            // sorted-ascending insertion
                kd[K_CROSS - 1] = d; kidx[K_CROSS - 1] = t0 + j;
#pragma unroll
                for (int p = K_CROSS - 1; p > 0; --p) {
                    const bool sw = kd[p] < kd[p - 1];
                    const float tv = kd[p]; const int ti = kidx[p];
                    kd[p]       = sw ? kd[p - 1]   : tv;
                    kidx[p]     = sw ? kidx[p - 1] : ti;
                    kd[p - 1]   = sw ? tv : kd[p - 1];
                    kidx[p - 1] = sw ? ti : kidx[p - 1];
                }
            }
        }
    }
    int* orow = knn + (base + n) * K_CROSS;
#pragma unroll
    for (int k = 0; k < K_CROSS; ++k) orow[k] = kidx[k];
}

// ---------- chamfer half-term: mean_n min_m |a_n - b_m|^2 (per-block partial)
__global__ __launch_bounds__(256)
void rowmin_mean_kernel(const float* __restrict__ Ap, const float* __restrict__ Bp,
                        float scale, float* __restrict__ partial)
{
    __shared__ float sx[256], sy[256], sz[256];
    __shared__ float red[256];
    const int tid = threadIdx.x;
    const int gid = blockIdx.x * 256 + tid;
    const int b = gid / N_, n = gid % N_;
    const size_t base = (size_t)b * N_;
    const float* arow = Ap + (base + n) * 3;
    const float ax = arow[0], ay = arow[1], az = arow[2];

    float m = 3.0e38f;
    for (int t0 = 0; t0 < N_; t0 += 256) {
        __syncthreads();
        const float* bp = Bp + (base + t0 + tid) * 3;
        sx[tid] = bp[0]; sy[tid] = bp[1]; sz[tid] = bp[2];
        __syncthreads();
#pragma unroll 4
        for (int j = 0; j < 256; ++j) {
            const float dx = ax - sx[j], dy = ay - sy[j], dz = az - sz[j];
            m = fminf(m, dx * dx + dy * dy + dz * dz);
        }
    }
    red[tid] = m;
    __syncthreads();
#pragma unroll
    for (int s = 128; s > 0; s >>= 1) {
        if (tid < s) red[tid] += red[tid + s];
        __syncthreads();
    }
    if (tid == 0) partial[blockIdx.x] = scale * red[0];
}

// ---------- neighbor smoothness loss (per-block partial) --------------------
__global__ __launch_bounds__(256)
void neighbor_loss_kernel(const float* __restrict__ pts, const int* __restrict__ knn,
                          const float* __restrict__ rec, float scale,
                          float* __restrict__ partial)
{
    __shared__ float red[256];
    const int tid = threadIdx.x;
    const int gid = blockIdx.x * 256 + tid;
    const int b = gid / N_, n = gid % N_;
    const size_t base = (size_t)b * N_;
    const float* p = pts + (base + n) * 3;
    const float* q = rec + (base + n) * 3;
    const float px = p[0], py = p[1], pz = p[2];
    const float qx = q[0], qy = q[1], qz = q[2];
    const int* nb = knn + (base + n) * K_CROSS;

    float acc = 0.f;
#pragma unroll
    for (int k = 1; k < K_CROSS; ++k) {           // entry 0 is the point itself
        const int j = nb[k];
        const float* pj = pts + (base + j) * 3;
        const float* qj = rec + (base + j) * 3;
        const float dx = px - pj[0], dy = py - pj[1], dz = pz - pj[2];
        const float sd = dx * dx + dy * dy + dz * dz;
        const float ex = __expf(-sd * INV_HEAT);
        const float ex0 = qx - qj[0], ey0 = qy - qj[1], ez0 = qz - qj[2];
        acc += ex * (ex0 * ex0 + ey0 * ey0 + ez0 * ez0);
    }
    red[tid] = acc;
    __syncthreads();
#pragma unroll
    for (int s = 128; s > 0; s >>= 1) {
        if (tid < s) red[tid] += red[tid + s];
        __syncthreads();
    }
    if (tid == 0) partial[blockIdx.x] = scale * red[0];
}

// ---------- deterministic fixed-order final sum -----------------------------
__global__ void finalize_loss_kernel(const float* __restrict__ partial, int n,
                                     float* __restrict__ loss)
{
    if (threadIdx.x == 0 && blockIdx.x == 0) {
        float s = 0.f;
        for (int i = 0; i < n; ++i) s += partial[i];
        *loss = s;
    }
}

} // anonymous namespace

extern "C" void kernel_launch(void* const* d_in, const int* in_sizes, int n_in,
                              void* d_out, int out_size, void* d_ws, size_t ws_size,
                              hipStream_t stream)
{
    (void)in_sizes; (void)n_in; (void)out_size; (void)ws_size;
    const float* src      = (const float*)d_in[0];
    const float* tgt      = (const float*)d_in[1];
    /* d_in[2] = gt : unused by the reference computation */
    const float* src_feat = (const float*)d_in[3];
    const float* tgt_feat = (const float*)d_in[4];

    // workspace carve-up (~5.2 MB total)
    char* ws = (char*)d_ws;
    size_t o = 0;
    _Float16* srcN = (_Float16*)(ws + o); o += (size_t)B_ * N_ * F_ * sizeof(_Float16);
    _Float16* tgtN = (_Float16*)(ws + o); o += (size_t)B_ * N_ * F_ * sizeof(_Float16);
    float* tcr = (float*)(ws + o); o += (size_t)B_ * N_ * 3 * sizeof(float);
    float* ssr = (float*)(ws + o); o += (size_t)B_ * N_ * 3 * sizeof(float);
    float* tsr = (float*)(ws + o); o += (size_t)B_ * N_ * 3 * sizeof(float);
    int* knnS = (int*)(ws + o); o += (size_t)B_ * N_ * K_CROSS * sizeof(int);
    int* knnT = (int*)(ws + o); o += (size_t)B_ * N_ * K_CROSS * sizeof(int);
    float* part = (float*)(ws + o);               // 10 regions x 32 block partials

    float* scr  = (float*)d_out;                  // source_cross_recon (output 0)
    float* loss = scr + (size_t)B_ * N_ * 3;      // scalar loss (output 1)

    const dim3 b256(256);
    const dim3 gNorm(B_ * N_ / 8);
    normalize_f16_kernel<<<gNorm, b256, 0, stream>>>(src_feat, srcN);
    normalize_f16_kernel<<<gNorm, b256, 0, stream>>>(tgt_feat, tgtN);

    const dim3 gW(B_ * (N_ / 32)), bW(32);
    // rows of P (src queries) -> weights over tgt -> target_cross_recon
    wmma_topk_recon<10, 0><<<gW, bW, 0, stream>>>(srcN, tgtN, tgt, tcr);
    // rows of P^T (tgt queries) -> weights over src -> source_cross_recon
    wmma_topk_recon<10, 0><<<gW, bW, 0, stream>>>(tgtN, srcN, src, scr);
    // self recon: keep 11, drop the self-match
    wmma_topk_recon<11, 1><<<gW, bW, 0, stream>>>(srcN, srcN, src, ssr);
    wmma_topk_recon<11, 1><<<gW, bW, 0, stream>>>(tgtN, tgtN, tgt, tsr);

    const dim3 g32(B_ * N_ / 256);
    knn10_kernel<<<g32, b256, 0, stream>>>(src, knnS);
    knn10_kernel<<<g32, b256, 0, stream>>>(tgt, knnT);

    // cd_t(a,b) = rowmin_mean(a,b) + rowmin_mean(b,a); batch-mean -> 1/(B*N)
    const float sCD = 1.0f / (float)(B_ * N_);
    rowmin_mean_kernel<<<g32, b256, 0, stream>>>(src, scr, sCD, part + 0 * 32);
    rowmin_mean_kernel<<<g32, b256, 0, stream>>>(scr, src, sCD, part + 1 * 32);
    rowmin_mean_kernel<<<g32, b256, 0, stream>>>(tgt, tcr, sCD, part + 2 * 32);
    rowmin_mean_kernel<<<g32, b256, 0, stream>>>(tcr, tgt, sCD, part + 3 * 32);
    rowmin_mean_kernel<<<g32, b256, 0, stream>>>(src, ssr, sCD, part + 4 * 32);
    rowmin_mean_kernel<<<g32, b256, 0, stream>>>(ssr, src, sCD, part + 5 * 32);
    rowmin_mean_kernel<<<g32, b256, 0, stream>>>(tgt, tsr, sCD, part + 6 * 32);
    rowmin_mean_kernel<<<g32, b256, 0, stream>>>(tsr, tgt, sCD, part + 7 * 32);

    const float sNB = 1.0f / (float)(B_ * N_ * (K_CROSS - 1));
    neighbor_loss_kernel<<<g32, b256, 0, stream>>>(src, knnS, tcr, sNB, part + 8 * 32);
    neighbor_loss_kernel<<<g32, b256, 0, stream>>>(tgt, knnT, scr, sNB, part + 9 * 32);

    finalize_loss_kernel<<<1, 1, 0, stream>>>(part, 10 * 32, loss);
}
